// GroupAttention_70944269795893
// MI455X (gfx1250) — matrix operand order repaired and checked by reading, not compiled, not run on hardware
//
#include <hip/hip_runtime.h>
#include <hip/hip_bf16.h>

#define HDIM 1024
#define BB   8
#define SS   1024

typedef _Float16 v16h __attribute__((ext_vector_type(16)));
typedef _Float16 v8h  __attribute__((ext_vector_type(8)));
typedef float    v8f  __attribute__((ext_vector_type(8)));

// ---------------------------------------------------------------------------
// 1) LayerNorm -> f16 (one block per row, 256 threads x 4 floats)
// ---------------------------------------------------------------------------
__global__ __launch_bounds__(256) void ga_ln_f16(const float* __restrict__ x,
                                                 const float* __restrict__ lw,
                                                 const float* __restrict__ lb,
                                                 _Float16* __restrict__ out)
{
    const int row = blockIdx.x;
    const int tid = threadIdx.x;
    const float4 v = ((const float4*)(x + (size_t)row * HDIM))[tid];
    float s  = v.x + v.y + v.z + v.w;
    float s2 = v.x * v.x + v.y * v.y + v.z * v.z + v.w * v.w;
    for (int off = 16; off > 0; off >>= 1) {
        s  += __shfl_down(s,  off);
        s2 += __shfl_down(s2, off);
    }
    __shared__ float red[16];
    __shared__ float sh_mu, sh_rs;
    const int wave = tid >> 5, lane = tid & 31;
    if (lane == 0) { red[wave] = s; red[8 + wave] = s2; }
    __syncthreads();
    if (tid == 0) {
        float ts = 0.f, ts2 = 0.f;
        for (int i = 0; i < 8; ++i) { ts += red[i]; ts2 += red[8 + i]; }
        const float mu = ts * (1.0f / HDIM);
        sh_mu = mu;
        sh_rs = rsqrtf(ts2 * (1.0f / HDIM) - mu * mu + 1e-5f);
    }
    __syncthreads();
    const float mu = sh_mu, rs = sh_rs;
    const float4 w4 = ((const float4*)lw)[tid];
    const float4 b4 = ((const float4*)lb)[tid];
    _Float16* o = out + (size_t)row * HDIM + tid * 4;
    o[0] = (_Float16)((v.x - mu) * rs * w4.x + b4.x);
    o[1] = (_Float16)((v.y - mu) * rs * w4.y + b4.y);
    o[2] = (_Float16)((v.z - mu) * rs * w4.z + b4.z);
    o[3] = (_Float16)((v.w - mu) * rs * w4.w + b4.w);
}

// ---------------------------------------------------------------------------
// 2) Weight fp32 -> f16
// ---------------------------------------------------------------------------
__global__ __launch_bounds__(256) void ga_cvt16(const float* __restrict__ wq,
                                                const float* __restrict__ wk,
                                                _Float16* __restrict__ oq,
                                                _Float16* __restrict__ ok)
{
    const int i = blockIdx.x * 256 + threadIdx.x;
    oq[i] = (_Float16)wq[i];
    ok[i] = (_Float16)wk[i];
}

// ---------------------------------------------------------------------------
// 3) WMMA GEMM: Q = h@Wq^T + bq ; K = h@Wk^T + bk (f16 in, f32 accumulate)
//    Per wave: 16(M) x 64(N) strip for BOTH matrices; A fragment reused 8x.
//    grid = (8192/16/8, 1024/64) = (64, 16), block = 256 (8 waves)
// ---------------------------------------------------------------------------
__global__ __launch_bounds__(256) void ga_gemm_qk(const _Float16* __restrict__ h16,
                                                  const _Float16* __restrict__ wq16,
                                                  const _Float16* __restrict__ wk16,
                                                  const float* __restrict__ bq,
                                                  const float* __restrict__ bk,
                                                  float* __restrict__ Q,
                                                  float* __restrict__ K)
{
    const int wave = threadIdx.x >> 5;
    const int lane = threadIdx.x & 31;
    const int hs   = lane >> 4;        // half-wave select
    const int l16  = lane & 15;
    const int m0 = (blockIdx.x * 8 + wave) * 16;
    const int n0 = blockIdx.y * 64;

    // A: row m0+l16, K chunks {0..15} (hs=0) / {16..31} (hs=1)
    const _Float16* arow = h16 + (size_t)(m0 + l16) * HDIM + hs * 16;

    v8f accq[4], acck[4];
    {
        v8f z = {};
        #pragma unroll
        for (int i = 0; i < 4; ++i) { accq[i] = z; acck[i] = z; }
    }

    for (int kk = 0; kk < HDIM; kk += 32) {
        v8h a0 = *(const v8h*)(arow + kk);
        v8h a1 = *(const v8h*)(arow + kk + 8);
        v16h a = __builtin_shufflevector(a0, a1,
                 0,1,2,3,4,5,6,7,8,9,10,11,12,13,14,15);
        const int kb = kk + hs * 16;   // B: lanes<16 -> K=kk..kk+15, lanes>=16 -> kk+16..kk+31
        #pragma unroll
        for (int nt = 0; nt < 4; ++nt) {
            const size_t roff = (size_t)(n0 + nt * 16 + l16) * HDIM + kb;
            v16h bqf = *(const v16h*)(wq16 + roff);
            accq[nt] = __builtin_amdgcn_wmma_f32_16x16x32_f16(
                false, a, false, bqf, (short)0, accq[nt], false, false);
            v16h bkf = *(const v16h*)(wk16 + roff);
            acck[nt] = __builtin_amdgcn_wmma_f32_16x16x32_f16(
                false, a, false, bkf, (short)0, acck[nt], false, false);
        }
    }

    // D layout: elem v -> row m0 + v + 8*hs, col n0 + nt*16 + l16
    const int rbase = m0 + hs * 8;
    #pragma unroll
    for (int nt = 0; nt < 4; ++nt) {
        const int col = n0 + nt * 16 + l16;
        const float bqv = bq[col], bkv = bk[col];
        #pragma unroll
        for (int v = 0; v < 8; ++v) {
            const size_t idx = (size_t)(rbase + v) * HDIM + col;
            Q[idx] = accq[nt][v] + bqv;
            K[idx] = acck[nt][v] + bkv;
        }
    }
}

// ---------------------------------------------------------------------------
// 4) Neighbor dots + 2-entry softmax -> raw_up[b,s]=na[s,s+1], raw_dn[b,s]=na[s,s-1]
// ---------------------------------------------------------------------------
__global__ __launch_bounds__(256) void ga_nbr(const float* __restrict__ Q,
                                              const float* __restrict__ K,
                                              const int* __restrict__ am,
                                              float* __restrict__ rup,
                                              float* __restrict__ rdn)
{
    const int bs = blockIdx.x;
    const int s  = bs & (SS - 1);
    const int tid = threadIdx.x;
    const float4 q = ((const float4*)(Q + (size_t)bs * HDIM))[tid];
    float du = 0.f, dd = 0.f;
    if (s < SS - 1) {
        const float4 k = ((const float4*)(K + (size_t)(bs + 1) * HDIM))[tid];
        du = q.x * k.x + q.y * k.y + q.z * k.z + q.w * k.w;
    }
    if (s > 0) {
        const float4 k = ((const float4*)(K + (size_t)(bs - 1) * HDIM))[tid];
        dd = q.x * k.x + q.y * k.y + q.z * k.z + q.w * k.w;
    }
    for (int off = 16; off > 0; off >>= 1) {
        du += __shfl_down(du, off);
        dd += __shfl_down(dd, off);
    }
    __shared__ float red[16];
    const int wave = tid >> 5, lane = tid & 31;
    if (lane == 0) { red[wave] = du; red[8 + wave] = dd; }
    __syncthreads();
    if (tid == 0) {
        float tu = 0.f, td = 0.f;
        for (int i = 0; i < 8; ++i) { tu += red[i]; td += red[8 + i]; }
        const size_t rowoff = (size_t)bs * SS;
        float xu = -1e9f, xd = -1e9f;
        if (s < SS - 1 && am[rowoff + s + 1] != 0) xu = tu * (1.0f / HDIM);
        if (s > 0      && am[rowoff + s - 1] != 0) xd = td * (1.0f / HDIM);
        const float mx  = fmaxf(xu, xd);
        const float eu  = expf(xu - mx);   // masked/edge terms underflow to exact 0
        const float ed  = expf(xd - mx);
        const float inv = 1.0f / (eu + ed);
        rup[bs] = (s < SS - 1) ? eu * inv : 0.f;
        rdn[bs] = (s > 0)      ? ed * inv : 0.f;
    }
}

// ---------------------------------------------------------------------------
// 5) Per-batch: L[s]=log(na_final[s,s+1]+1e-9); SL = exclusive prefix sum
// ---------------------------------------------------------------------------
__global__ __launch_bounds__(1024) void ga_scan(const float* __restrict__ prev,
                                                const float* __restrict__ rup,
                                                const float* __restrict__ rdn,
                                                float* __restrict__ SL)
{
    const int b = blockIdx.x;
    const int s = threadIdx.x;
    const int bs = b * SS + s;
    float L = 0.f;
    if (s < SS - 1) {
        const float p  = prev[(size_t)bs * SS + s + 1];
        const float na = p + (1.f - p) * sqrtf(rup[bs] * rdn[bs + 1] + 1e-9f);
        L = logf(na + 1e-9f);
    }
    __shared__ float buf[SS];
    buf[s] = L;
    __syncthreads();
    for (int off = 1; off < SS; off <<= 1) {   // Hillis-Steele inclusive scan
        const float t = (s >= off) ? buf[s - off] : 0.f;
        __syncthreads();
        buf[s] += t;
        __syncthreads();
    }
    SL[bs] = (s == 0) ? 0.f : buf[s - 1];
}

// ---------------------------------------------------------------------------
// 6) Dense outputs: na_out and C (bandwidth-bound; float4 I/O, SL row in LDS)
// ---------------------------------------------------------------------------
__global__ __launch_bounds__(256) void ga_final(const float* __restrict__ prev,
                                                const float* __restrict__ rup,
                                                const float* __restrict__ rdn,
                                                const float* __restrict__ SL,
                                                float* __restrict__ C,
                                                float* __restrict__ NA)
{
    const int bs = blockIdx.x;
    const int b  = bs >> 10;
    const int s  = bs & (SS - 1);
    const size_t rowoff = (size_t)bs * SS;

    __shared__ float slrow[SS];
    ((float4*)slrow)[threadIdx.x] = ((const float4*)(SL + (size_t)b * SS))[threadIdx.x];
    __syncthreads();

    const float SLs   = slrow[s];
    const float rup_s = rup[bs];
    const float rdn_s = rdn[bs];
    const float rup_m = (s > 0)      ? rup[bs - 1] : 0.f;
    const float rdn_p = (s < SS - 1) ? rdn[bs + 1] : 0.f;

    const float4 p4 = ((const float4*)(prev + rowoff))[threadIdx.x];
    const float pv[4] = { p4.x, p4.y, p4.z, p4.w };
    float na4[4], c4[4];
    const int t0 = threadIdx.x * 4;
    #pragma unroll
    for (int i = 0; i < 4; ++i) {
        const int t = t0 + i;
        float prod = 0.f;
        if (t == s + 1)      prod = rup_s * rdn_p;
        else if (t == s - 1) prod = rdn_s * rup_m;
        const float p  = pv[i];
        const float na = p + (1.f - p) * sqrtf(prod + 1e-9f);
        float c;
        if (t == s) c = na;                          // diagonal gets na[s,s]
        else {
            const float slt = slrow[t];
            c = expf((t > s) ? (slt - SLs) : (SLs - slt)) + 1e-9f;
        }
        na4[i] = na; c4[i] = c;
    }
    ((float4*)(NA + rowoff))[threadIdx.x] = make_float4(na4[0], na4[1], na4[2], na4[3]);
    ((float4*)(C  + rowoff))[threadIdx.x] = make_float4(c4[0], c4[1], c4[2], c4[3]);
}

// ---------------------------------------------------------------------------
extern "C" void kernel_launch(void* const* d_in, const int* in_sizes, int n_in,
                              void* d_out, int out_size, void* d_ws, size_t ws_size,
                              hipStream_t stream)
{
    (void)in_sizes; (void)n_in; (void)out_size; (void)ws_size;
    const float* hidden = (const float*)d_in[0];
    const int*   amask  = (const int*)  d_in[1];
    const float* prev   = (const float*)d_in[2];
    const float* ln_w   = (const float*)d_in[3];
    const float* ln_b   = (const float*)d_in[4];
    const float* Wq     = (const float*)d_in[5];
    const float* bq     = (const float*)d_in[6];
    const float* Wk     = (const float*)d_in[7];
    const float* bk     = (const float*)d_in[8];

    const size_t nBS = (size_t)BB * SS;            // 8192 rows
    char* ws = (char*)d_ws;
    _Float16* h16  = (_Float16*)ws;  ws += nBS * HDIM * sizeof(_Float16);          // 16 MB
    _Float16* wq16 = (_Float16*)ws;  ws += (size_t)HDIM * HDIM * sizeof(_Float16); //  2 MB
    _Float16* wk16 = (_Float16*)ws;  ws += (size_t)HDIM * HDIM * sizeof(_Float16); //  2 MB
    float* Qb  = (float*)ws;         ws += nBS * HDIM * sizeof(float);             // 32 MB
    float* Kb  = (float*)ws;         ws += nBS * HDIM * sizeof(float);             // 32 MB
    float* rup = (float*)ws;         ws += nBS * sizeof(float);
    float* rdn = (float*)ws;         ws += nBS * sizeof(float);
    float* SLb = (float*)ws;         ws += nBS * sizeof(float);

    float* Cout  = (float*)d_out;
    float* NAout = (float*)d_out + (size_t)BB * SS * SS;

    ga_ln_f16 <<<(unsigned)nBS, 256, 0, stream>>>(hidden, ln_w, ln_b, h16);
    ga_cvt16  <<<(HDIM * HDIM) / 256, 256, 0, stream>>>(Wq, Wk, wq16, wk16);
    ga_gemm_qk<<<dim3(64, 16), 256, 0, stream>>>(h16, wq16, wk16, bq, bk, Qb, Kb);
    ga_nbr    <<<(unsigned)nBS, 256, 0, stream>>>(Qb, Kb, amask, rup, rdn);
    ga_scan   <<<BB, SS, 0, stream>>>(prev, rup, rdn, SLb);
    ga_final  <<<(unsigned)nBS, 256, 0, stream>>>(prev, rup, rdn, SLb, Cout, NAout);
}
